// InterModalityUpdate_89283780150024
// MI455X (gfx1250) — matrix operand order repaired and checked by reading, not compiled, not run on hardware
//
#include <hip/hip_runtime.h>
#include <hip/hip_bf16.h>

// ---------------------------------------------------------------------------
// InterModalityUpdate on MI455X (gfx1250).
// - All GEMMs + attention contractions on v_wmma_f32_16x16x32_f16.
// - Tile staging via GLOBAL_LOAD_ASYNC_TO_LDS_B128, double-buffered with
//   partial s_wait_asynccnt so DMA overlaps WMMA (ASYNCcnt completes in-order).
// - Softmax row reductions via DPP16 (no LDS round-trips).
// ---------------------------------------------------------------------------

typedef _Float16 h16;
typedef __attribute__((ext_vector_type(16))) _Float16 v16h;
typedef __attribute__((ext_vector_type(8)))  _Float16 v8h;
typedef __attribute__((ext_vector_type(8)))  float    v8f;

static constexpr int BATCH = 32;
static constexpr int NVQ   = 512;
static constexpr int DMODEL= 512;
static constexpr int H3    = 1536;
static constexpr int HEADS = 8;
static constexpr int HD    = 64;
static constexpr int MROWS = BATCH * NVQ;  // 16384

#ifndef USE_ASYNC_LDS
#define USE_ASYNC_LDS 1
#endif

// ---- async memory->LDS DMA (gfx1250 GLOBAL_LOAD_ASYNC_TO_LDS_B128) --------
__device__ inline void async_ld_b128(void* lds_ptr, const void* gaddr) {
#if USE_ASYNC_LDS
  unsigned lds_off = (unsigned)(size_t)lds_ptr;   // generic->LDS: low 32 bits
  asm volatile("global_load_async_to_lds_b128 %0, %1, off"
               :: "v"(lds_off), "v"(gaddr) : "memory");
#else
  *(v8h*)lds_ptr = *(const v8h*)gaddr;
#endif
}
// wait until at most N async ops remain outstanding (older groups complete
// in-order, so N = size of the newest group releases everything older)
template <int N>
__device__ inline void wait_async_le() {
#if USE_ASYNC_LDS
  asm volatile("s_wait_asynccnt %0" :: "n"(N) : "memory");
#endif
}

__device__ inline v16h cat8(v8h a, v8h b) {
  return __builtin_shufflevector(a, b, 0,1,2,3,4,5,6,7,8,9,10,11,12,13,14,15);
}

// D = A*B + C, 16x16x32 f16 -> f32
__device__ inline v8f wmma_f16(v16h a, v16h b, v8f c) {
  return __builtin_amdgcn_wmma_f32_16x16x32_f16(
      false, a, false, b, (short)0, c, false, false);
}

// A fragment (16x32, M x K) from row-major [16][ld] tile. ld % 8 == 0.
__device__ inline v16h frag_a(const h16* base, int ld) {
  int lane = threadIdx.x & 31;
  int m = lane & 15, hh = lane >> 4;
  const h16* r = base + m * ld + 8 * hh;
  return cat8(*(const v8h*)(r), *(const v8h*)(r + 16));
}

// B fragment (32x16, K x N) from B^T stored row-major [N][ld]. ld % 8 == 0.
__device__ inline v16h frag_b(const h16* base, int ld) {
  int lane = threadIdx.x & 31;
  int n = lane & 15, hh = lane >> 4;
  const h16* r = base + n * ld + 8 * hh;
  return cat8(*(const v8h*)r, *(const v8h*)(r + 16));
}

// ---- DPP16 butterfly reductions over each 16-lane half ---------------------
template <int CTRL>
__device__ inline float dpp_f(float x) {
  int xi = __float_as_int(x);
  int r  = __builtin_amdgcn_update_dpp(xi, xi, CTRL, 0xF, 0xF, true);
  return __int_as_float(r);
}
__device__ inline float red_max16(float x) {
  x = fmaxf(x, dpp_f<0xB1>(x));   // quad_perm xor1
  x = fmaxf(x, dpp_f<0x4E>(x));   // quad_perm xor2
  x = fmaxf(x, dpp_f<0x141>(x));  // row_half_mirror
  x = fmaxf(x, dpp_f<0x140>(x));  // row_mirror
  return x;
}
__device__ inline float red_sum16(float x) {
  x += dpp_f<0xB1>(x);
  x += dpp_f<0x4E>(x);
  x += dpp_f<0x141>(x);
  x += dpp_f<0x140>(x);
  return x;
}

// ---------------------------------------------------------------------------
__global__ void cvt_kernel(const float* __restrict__ x,
                           h16* __restrict__ relu_out,
                           h16* __restrict__ cat_out, int total) {
  int i = blockIdx.x * blockDim.x + threadIdx.x;
  if (i >= total) return;
  float v = x[i];
  relu_out[i] = (h16)fmaxf(v, 0.0f);
  int row = i >> 9;
  int col = i & 511;
  cat_out[(size_t)row * 1024 + col] = (h16)v;
}

// W[K][N] f32 -> Wt[N][K] f16
__global__ void transpose_kernel(const float* __restrict__ W,
                                 h16* __restrict__ Wt, int K, int N) {
  int i = blockIdx.x * blockDim.x + threadIdx.x;
  if (i >= K * N) return;
  int n = i / K, k = i % K;
  Wt[i] = (h16)W[(size_t)k * N + n];
}

// ---------------------------------------------------------------------------
// Tiled WMMA GEMM: C[M,N] = A[M,K] @ Wt[N,K]^T + bias.
// 256 threads (8 waves), BM=128 x BN=128; wave tile 32x64 (8 WMMA/iter).
// Double-buffered async DMA staging: DMA for step s+1 overlaps WMMA of step s.
// ---------------------------------------------------------------------------
template <bool STORE_F32>
__global__ __launch_bounds__(256) void gemm_kernel(
    const h16* __restrict__ A, const h16* __restrict__ Wt,
    const float* __restrict__ bias, void* __restrict__ Cout,
    int M, int N, int K) {
  constexpr int BM = 128, BN = 128, KT = 32, LDT = KT + 8;
  __shared__ h16 sA[2][BM * LDT];
  __shared__ h16 sB[2][BN * LDT];

  int tid  = threadIdx.x;
  int wave = tid >> 5;
  int wm   = wave >> 1;   // 0..3  (32 rows each)
  int wn   = wave & 1;    // 0..1  (64 cols each)
  int m0   = blockIdx.x * BM;
  int n0   = blockIdx.y * BN;

  v8f zero = {};
  v8f acc[2][4];
  for (int i = 0; i < 2; i++)
    for (int j = 0; j < 4; j++) acc[i][j] = zero;

  int r = tid >> 1;            // 0..127
  int c = (tid & 1) * 16;      // 0 or 16

  // stage one 128x32 A tile + 128x32 B tile (4 async b128 per thread)
  auto stage = [&](int bufi, int k0) {
    const h16* ga = A + (size_t)(m0 + r) * K + k0 + c;
    async_ld_b128(&sA[bufi][r * LDT + c],     ga);
    async_ld_b128(&sA[bufi][r * LDT + c + 8], ga + 8);
    const h16* gb = Wt + (size_t)(n0 + r) * K + k0 + c;
    async_ld_b128(&sB[bufi][r * LDT + c],     gb);
    async_ld_b128(&sB[bufi][r * LDT + c + 8], gb + 8);
  };

  int nsteps = K / KT;
  stage(0, 0);
  for (int s = 0; s < nsteps; s++) {
    int buf = s & 1;
    if (s + 1 < nsteps) {
      stage(buf ^ 1, (s + 1) * KT);  // DMA next tile while computing this one
      wait_async_le<4>();            // newest group = 4 => current tile landed
    } else {
      wait_async_le<0>();
    }
    __syncthreads();

    v16h af[2], bf[4];
#pragma unroll
    for (int i = 0; i < 2; i++) af[i] = frag_a(&sA[buf][(wm * 32 + i * 16) * LDT], LDT);
#pragma unroll
    for (int j = 0; j < 4; j++) bf[j] = frag_b(&sB[buf][(wn * 64 + j * 16) * LDT], LDT);
#pragma unroll
    for (int i = 0; i < 2; i++)
#pragma unroll
      for (int j = 0; j < 4; j++) acc[i][j] = wmma_f16(af[i], bf[j], acc[i][j]);

    __syncthreads();  // all waves done reading before next stage overwrites
  }

  int lane = tid & 31;
  int cn = lane & 15, hh = lane >> 4;
#pragma unroll
  for (int i = 0; i < 2; i++)
#pragma unroll
    for (int j = 0; j < 4; j++) {
      int col = n0 + wn * 64 + j * 16 + cn;
      float bv = bias ? bias[col] : 0.0f;
#pragma unroll
      for (int e = 0; e < 8; e++) {
        int row = m0 + wm * 32 + i * 16 + e + 8 * hh;
        float val = acc[i][j][e] + bv;
        if (STORE_F32)
          ((float*)Cout)[(size_t)row * N + col] = val;
        else
          ((h16*)Cout)[(size_t)row * N + col] = (h16)val;
      }
    }
}

// ---------------------------------------------------------------------------
// Flash-attention: workgroup = (128 queries, head, batch); 8 waves x 16 q.
// K via async DMA, V transposed in LDS — both double-buffered so the next
// block streams in during softmax/WMMA of the current block.
// ---------------------------------------------------------------------------
__global__ __launch_bounds__(256) void attn_kernel(
    const h16* __restrict__ Qb, const h16* __restrict__ Kb,
    const h16* __restrict__ Vb, h16* __restrict__ Ob,
    int ldq, int ldk, int ldo, int nq, int nk, float scale) {
  constexpr int BK = 64, LD = 72;
  __shared__ h16 sK[2][BK * LD];     // [key][d]
  __shared__ h16 sVt[2][HD * LD];    // [d][key]
  __shared__ h16 sP[8 * 16 * LD];    // per-wave P [16 q][64 k]

  int tid = threadIdx.x, wave = tid >> 5, lane = tid & 31;
  int b = blockIdx.z, h = blockIdx.y;
  int q0 = blockIdx.x * 128 + wave * 16;

  const h16* Q  = Qb + ((size_t)b * nq + q0) * ldq + h * HD;
  const h16* Kp = Kb + ((size_t)b * nk) * ldk + h * HD;
  const h16* Vp = Vb + ((size_t)b * nk) * ldk + h * HD;

  // Q fragments held in registers across all key blocks
  v16h qf[2];
  {
    int m = lane & 15, hh = lane >> 4;
    const h16* rq = Q + (size_t)m * ldq;
#pragma unroll
    for (int kf = 0; kf < 2; kf++)
      qf[kf] = cat8(*(const v8h*)(rq + kf * 32 + 8 * hh),
                    *(const v8h*)(rq + kf * 32 + 16 + 8 * hh));
  }

  float mrow[8], srow[8];
  v8f zero = {};
  v8f oacc[4];
#pragma unroll
  for (int i = 0; i < 8; i++) { mrow[i] = -1e30f; srow[i] = 0.0f; }
#pragma unroll
  for (int j = 0; j < 4; j++) oacc[j] = zero;

  h16* myP = &sP[wave * 16 * LD];
  int cn = lane & 15, hh = lane >> 4;

  int rr = tid >> 2;          // key 0..63
  int cc = (tid & 3) * 16;    // d

  // stage one K/V block: K via 2 async b128, V loaded+transposed into sVt
  auto stage = [&](int bufi, int kb2) {
    const h16* g = Kp + (size_t)(kb2 + rr) * ldk + cc;
    async_ld_b128(&sK[bufi][rr * LD + cc],     g);
    async_ld_b128(&sK[bufi][rr * LD + cc + 8], g + 8);
    const h16* gv = Vp + (size_t)(kb2 + rr) * ldk + cc;
    v8h y0 = *(const v8h*)gv;
    v8h y1 = *(const v8h*)(gv + 8);
#pragma unroll
    for (int e = 0; e < 8; e++) {
      sVt[bufi][(cc + e) * LD + rr]     = y0[e];
      sVt[bufi][(cc + 8 + e) * LD + rr] = y1[e];
    }
  };

  int nblocks = nk / BK;
  stage(0, 0);
  for (int t = 0; t < nblocks; t++) {
    int buf = t & 1;
    if (t + 1 < nblocks) {
      stage(buf ^ 1, (t + 1) * BK);  // stream next block during compute
      wait_async_le<2>();            // newest group = 2 => current K landed
    } else {
      wait_async_le<0>();
    }
    __syncthreads();

    // S = Q*K^T : 16 queries x 64 keys (4 subtiles x 2 k-frags)
    v8f sacc[4];
#pragma unroll
    for (int nt = 0; nt < 4; nt++) {
      v8f a = zero;
#pragma unroll
      for (int kf = 0; kf < 2; kf++)
        a = wmma_f16(qf[kf], frag_b(&sK[buf][(nt * 16) * LD + kf * 32], LD), a);
      sacc[nt] = a;
    }

    // online softmax: row i+8*hh lives across the 16 lanes of this half
#pragma unroll
    for (int i = 0; i < 8; i++) {
      float bm = -1e30f;
#pragma unroll
      for (int nt = 0; nt < 4; nt++) bm = fmaxf(bm, sacc[nt][i] * scale);
      bm = red_max16(bm);
      float nm = fmaxf(mrow[i], bm);
      float corr = __expf(mrow[i] - nm);
      float psum = 0.0f;
#pragma unroll
      for (int nt = 0; nt < 4; nt++) {
        float p = __expf(sacc[nt][i] * scale - nm);
        sacc[nt][i] = p;
        psum += p;
      }
      psum = red_sum16(psum);
      srow[i] = srow[i] * corr + psum;
      mrow[i] = nm;
#pragma unroll
      for (int j = 0; j < 4; j++) oacc[j][i] *= corr;
    }

    // P (C-layout) -> per-wave LDS -> A-fragments
#pragma unroll
    for (int nt = 0; nt < 4; nt++)
#pragma unroll
      for (int i = 0; i < 8; i++)
        myP[(i + 8 * hh) * LD + nt * 16 + cn] = (h16)sacc[nt][i];

    v16h pf[2];
#pragma unroll
    for (int kf = 0; kf < 2; kf++) pf[kf] = frag_a(&myP[kf * 32], LD);
#pragma unroll
    for (int dt = 0; dt < 4; dt++)
#pragma unroll
      for (int kf = 0; kf < 2; kf++)
        oacc[dt] = wmma_f16(pf[kf], frag_b(&sVt[buf][(dt * 16) * LD + kf * 32], LD), oacc[dt]);

    __syncthreads();  // all waves done with buf before it is restaged
  }

  // normalize and store f16 into concat buffer
  h16* O = Ob + ((size_t)b * nq + q0) * ldo + h * HD;
#pragma unroll
  for (int dt = 0; dt < 4; dt++)
#pragma unroll
    for (int i = 0; i < 8; i++) {
      int row = i + 8 * hh;
      O[(size_t)row * ldo + dt * 16 + cn] = (h16)(oacc[dt][i] / srow[i]);
    }
}

// ---------------------------------------------------------------------------
extern "C" void kernel_launch(void* const* d_in, const int* in_sizes, int n_in,
                              void* d_out, int out_size, void* d_ws, size_t ws_size,
                              hipStream_t stream) {
  const float* v     = (const float*)d_in[0];
  const float* q     = (const float*)d_in[1];
  const float* Wvlin = (const float*)d_in[2];
  const float* bvlin = (const float*)d_in[3];
  const float* Wqlin = (const float*)d_in[4];
  const float* bqlin = (const float*)d_in[5];
  const float* Wvout = (const float*)d_in[6];
  const float* bvout = (const float*)d_in[7];
  const float* Wqout = (const float*)d_in[8];
  const float* bqout = (const float*)d_in[9];

  char* w = (char*)d_ws;
  auto alloc = [&](size_t nelem) -> h16* {
    h16* p = (h16*)w;
    w += ((nelem * sizeof(h16) + 255) / 256) * 256;
    return p;
  };
  h16* vh   = alloc((size_t)MROWS * DMODEL);   // relu(v) f16
  h16* qh   = alloc((size_t)MROWS * DMODEL);   // relu(q) f16
  h16* vcat = alloc((size_t)MROWS * 1024);     // [v | v_update] f16
  h16* qcat = alloc((size_t)MROWS * 1024);     // [q | q_update] f16
  h16* vtr  = alloc((size_t)MROWS * H3);       // v_trans f16
  h16* qtr  = alloc((size_t)MROWS * H3);       // q_trans f16
  h16* Wvl  = alloc((size_t)H3 * DMODEL);      // W_vlin^T
  h16* Wql  = alloc((size_t)H3 * DMODEL);      // W_qlin^T
  h16* Wvo  = alloc((size_t)DMODEL * 1024);    // W_vout^T
  h16* Wqo  = alloc((size_t)DMODEL * 1024);    // W_qout^T

  int total = MROWS * DMODEL;
  cvt_kernel<<<(total + 255) / 256, 256, 0, stream>>>(v, vh, vcat, total);
  cvt_kernel<<<(total + 255) / 256, 256, 0, stream>>>(q, qh, qcat, total);

  int wlin = H3 * DMODEL, wout = DMODEL * 1024;
  transpose_kernel<<<(wlin + 255) / 256, 256, 0, stream>>>(Wvlin, Wvl, DMODEL, H3);
  transpose_kernel<<<(wlin + 255) / 256, 256, 0, stream>>>(Wqlin, Wql, DMODEL, H3);
  transpose_kernel<<<(wout + 255) / 256, 256, 0, stream>>>(Wvout, Wvo, 1024, DMODEL);
  transpose_kernel<<<(wout + 255) / 256, 256, 0, stream>>>(Wqout, Wqo, 1024, DMODEL);

  dim3 blk(256);

  // input linears: [16384,512] @ [512,1536]
  dim3 g1(MROWS / 128, H3 / 128);
  gemm_kernel<false><<<g1, blk, 0, stream>>>(vh, Wvl, bvlin, vtr, MROWS, H3, DMODEL);
  gemm_kernel<false><<<g1, blk, 0, stream>>>(qh, Wql, bqlin, qtr, MROWS, H3, DMODEL);

  // attention: Q=*_q (cols 512..1023), K=*_k (cols 0..511), V=*_v (cols 1024..1535)
  dim3 ga(NVQ / 128, HEADS, BATCH);
  float scale = 0.125f;  // 1/sqrt(64)
  attn_kernel<<<ga, blk, 0, stream>>>(vtr + 512, qtr, qtr + 1024, vcat + 512,
                                      H3, H3, 1024, NVQ, NVQ, scale);
  attn_kernel<<<ga, blk, 0, stream>>>(qtr + 512, vtr, vtr + 1024, qcat + 512,
                                      H3, H3, 1024, NVQ, NVQ, scale);

  // output projections: [16384,1024] @ [1024,512] + bias -> f32 d_out
  dim3 g2(MROWS / 128, DMODEL / 128);
  float* out_v = (float*)d_out;
  float* out_q = out_v + (size_t)MROWS * DMODEL;
  gemm_kernel<true><<<g2, blk, 0, stream>>>(vcat, Wvo, bvout, out_v, MROWS, DMODEL, 1024);
  gemm_kernel<true><<<g2, blk, 0, stream>>>(qcat, Wqo, bqout, out_q, MROWS, DMODEL, 1024);
}
